// Net_34737695490170
// MI455X (gfx1250) — matrix operand order repaired and checked by reading, not compiled, not run on hardware
//
#include <hip/hip_runtime.h>
#include <stdint.h>

typedef __attribute__((ext_vector_type(16))) _Float16 v16h;
typedef __attribute__((ext_vector_type(8)))  _Float16 v8h;
typedef __attribute__((ext_vector_type(8)))  float    v8f;

#define TPB 256

// ---------------- helpers ----------------
__device__ __forceinline__ uint32_t pcg_hash(uint32_t v){
  v = v*747796405u + 2891336453u;
  uint32_t w = ((v >> ((v>>28u)+4u)) ^ v) * 277803737u;
  return (w>>22u)^w;
}
__device__ __forceinline__ float u01(uint32_t h){ return (float)(h>>8) * (1.0f/16777216.0f); }

// ---------------- degree / norm ----------------
__global__ void k_deg_scatter(const int* __restrict__ dst, const float* __restrict__ w,
                              float* __restrict__ deg, int Etot){
  int e = blockIdx.x*blockDim.x + threadIdx.x;
  if(e < Etot) atomicAdd(&deg[dst[e]], w[e]);
}
__global__ void k_dinv(const float* __restrict__ deg, float* __restrict__ dinv, int n){
  int i = blockIdx.x*blockDim.x + threadIdx.x;
  if(i < n) dinv[i] = rsqrtf(deg[i] + 1.0f);
}
__global__ void k_enorm(const int* __restrict__ src, const int* __restrict__ dst,
                        const float* __restrict__ w, const float* __restrict__ dinv,
                        float* __restrict__ enorm, int Etot){
  int e = blockIdx.x*blockDim.x + threadIdx.x;
  if(e < Etot) enorm[e] = dinv[src[e]] * w[e] * dinv[dst[e]];
}

// ---------------- small dense matmul (row-major), optional bias ----------------
__global__ void k_mm_bias(const float* __restrict__ h, const float* __restrict__ W,
                          const float* __restrict__ bias, float* __restrict__ out,
                          int n, int din, int dout){
  long idx = (long)blockIdx.x*blockDim.x + threadIdx.x;
  if(idx >= (long)n*dout) return;
  int i = (int)(idx/dout), j = (int)(idx - (long)i*dout);
  float s = bias ? bias[j] : 0.0f;
  for(int k = 0; k < din; ++k) s += h[(long)i*din + k] * W[(long)k*dout + j];
  out[idx] = s;
}

// ---------------- GCN scatter (sym-normalized) ----------------
__global__ void k_gcn_scatter(const int* __restrict__ src, const int* __restrict__ dst,
                              const float* __restrict__ enorm, const float* __restrict__ hw,
                              float* __restrict__ acc, int Etot, int d){
  long t = (long)blockIdx.x*blockDim.x + threadIdx.x;
  if(t >= (long)Etot*d) return;
  int e = (int)(t/d), j = (int)(t - (long)e*d);
  atomicAdd(&acc[(long)dst[e]*d + j], hw[(long)src[e]*d + j] * enorm[e]);
}
__global__ void k_gcn_finish(const float* __restrict__ acc, const float* __restrict__ hw,
                             const float* __restrict__ dinv, const float* __restrict__ bias,
                             float* __restrict__ out, int n, int d){
  long idx = (long)blockIdx.x*blockDim.x + threadIdx.x;
  if(idx >= (long)n*d) return;
  int i = (int)(idx/d), j = (int)(idx - (long)i*d);
  float sw = dinv[i]*dinv[i];
  out[idx] = acc[idx] + hw[idx]*sw + bias[j];
}

// ---------------- batchnorm (training-mode batch stats) ----------------
__global__ void k_colstat(const float* __restrict__ h, float* __restrict__ stat, int n, int d){
  int j = blockIdx.x;                 // one block per column
  float s = 0.f, s2 = 0.f;
  for(int i = threadIdx.x; i < n; i += blockDim.x){
    float v = h[(long)i*d + j]; s += v; s2 += v*v;
  }
  __shared__ float ls[TPB], lq[TPB];
  ls[threadIdx.x] = s; lq[threadIdx.x] = s2; __syncthreads();
  for(int o = TPB/2; o > 0; o >>= 1){
    if(threadIdx.x < o){ ls[threadIdx.x] += ls[threadIdx.x+o]; lq[threadIdx.x] += lq[threadIdx.x+o]; }
    __syncthreads();
  }
  if(threadIdx.x == 0){
    float m = ls[0]/(float)n;
    stat[2*j] = m; stat[2*j+1] = lq[0]/(float)n - m*m;
  }
}
__global__ void k_bn_apply(float* __restrict__ h, const float* __restrict__ stat,
                           const float* __restrict__ g, const float* __restrict__ bb,
                           int n, int d){
  long idx = (long)blockIdx.x*blockDim.x + threadIdx.x;
  if(idx >= (long)n*d) return;
  int j = (int)(idx % d);
  h[idx] = g[j]*(h[idx] - stat[2*j]) * rsqrtf(stat[2*j+1] + 1e-5f) + bb[j];
}

// ---------------- per-graph max pooling over rows ----------------
__global__ void k_pool(const float* __restrict__ a, const float* __restrict__ b,
                       const float* __restrict__ c, float* __restrict__ out,
                       int B, int rows, int d){
  int idx = blockIdx.x*blockDim.x + threadIdx.x;
  if(idx >= B*3*d) return;
  int bg = idx/(3*d), f = idx%(3*d);
  const float* h = (f < d) ? a : ((f < 2*d) ? b : c);
  int j = f % d;
  float m = -1e30f;
  for(int i = 0; i < rows; ++i) m = fmaxf(m, h[((long)bg*rows + i)*d + j]);
  out[idx] = m;
}

// ---------------- Belief Propagation ----------------
__global__ void k_bp_c(const float* __restrict__ betas, int mi, const float* __restrict__ w,
                       float* __restrict__ c, int Etot){
  int e = blockIdx.x*blockDim.x + threadIdx.x;
  if(e < Etot) c[e] = __expf(betas[mi]*w[e]) - 1.0f;
}
__global__ void k_bp_init(float* __restrict__ mu, int Etot, int q, int mi){
  int e = blockIdx.x*blockDim.x + threadIdx.x;
  if(e >= Etot) return;
  float v[8]; float mx = -1e30f;
  for(int j = 0; j < q; ++j){
    uint32_t s = (uint32_t)e*2654435761u ^ ((uint32_t)j*40503u) ^ ((uint32_t)mi*668265263u);
    float z = (u01(pcg_hash(s)) + u01(pcg_hash(s ^ 0x9E3779B9u)) + u01(pcg_hash(s ^ 0x85EBCA6Bu)) - 1.5f)*2.0f;
    v[j] = 0.1f*z; mx = fmaxf(mx, v[j]);
  }
  float den = 0.f;
  for(int j = 0; j < q; ++j){ v[j] = __expf(v[j]-mx); den += v[j]; }
  float inv = 1.0f/den;
  for(int j = 0; j < q; ++j) mu[(long)e*q + j] = v[j]*inv;
}
__global__ void k_bp_tfld(const int* __restrict__ dst, const float* __restrict__ c,
                          const float* __restrict__ mu, float* __restrict__ t,
                          float* __restrict__ fld, int Etot, int q){
  int e = blockIdx.x*blockDim.x + threadIdx.x;
  if(e >= Etot) return;
  float ce = c[e]; int d2 = dst[e];
  for(int j = 0; j < q; ++j){
    float tv = log1pf(ce*mu[(long)e*q + j]);
    t[(long)e*q + j] = tv;
    atomicAdd(&fld[(long)d2*q + j], tv);
  }
}
__global__ void k_bp_update(const int* __restrict__ src, const float* __restrict__ t,
                            const float* __restrict__ fld, float* __restrict__ mu,
                            int Etot, int q){
  int e = blockIdx.x*blockDim.x + threadIdx.x;
  if(e >= Etot) return;
  int E = Etot >> 1;
  int re = (e < E) ? (e + E) : (e - E);
  int s2 = src[e];
  float v[8]; float mx = -1e30f;
  for(int j = 0; j < q; ++j){ v[j] = fld[(long)s2*q + j] - t[(long)re*q + j]; mx = fmaxf(mx, v[j]); }
  float den = 0.f;
  for(int j = 0; j < q; ++j){ v[j] = __expf(v[j]-mx); den += v[j]; }
  float inv = 1.0f/den;
  for(int j = 0; j < q; ++j) mu[(long)e*q + j] = v[j]*inv;
}
__global__ void k_bp_marg(const float* __restrict__ fld, float* __restrict__ s1,
                          int n, int q, int off){
  int i = blockIdx.x*blockDim.x + threadIdx.x;
  if(i >= n) return;
  float v[8]; float mx = -1e30f;
  for(int j = 0; j < q; ++j){ v[j] = fld[(long)i*q + j]; mx = fmaxf(mx, v[j]); }
  float den = 0.f;
  for(int j = 0; j < q; ++j){ v[j] = __expf(v[j]-mx); den += v[j]; }
  float inv = 1.0f/den;
  for(int j = 0; j < q; ++j) s1[(long)i*35 + off + j] = v[j]*inv;
}

// ---------------- row softmax (d=100) ----------------
__global__ void k_rowsoftmax(float* __restrict__ h, int n, int d){
  int i = blockIdx.x*blockDim.x + threadIdx.x;
  if(i >= n) return;
  float mx = -1e30f;
  for(int j = 0; j < d; ++j) mx = fmaxf(mx, h[(long)i*d + j]);
  float den = 0.f;
  for(int j = 0; j < d; ++j){ float e = __expf(h[(long)i*d + j]-mx); h[(long)i*d + j] = e; den += e; }
  float inv = 1.0f/den;
  for(int j = 0; j < d; ++j) h[(long)i*d + j] *= inv;
}

// ---------------- losses ----------------
__global__ void k_ml(const int* __restrict__ src, const int* __restrict__ dst,
                     const float* __restrict__ sg, float* __restrict__ accum, int Etot){
  int e = blockIdx.x*blockDim.x + threadIdx.x;
  float s = 0.f;
  if(e < Etot){
    const float* a = &sg[(long)src[e]*100];
    const float* b = &sg[(long)dst[e]*100];
    for(int j = 0; j < 100; ++j) s += a[j]*b[j];
  }
  __shared__ float red[TPB];
  red[threadIdx.x] = s; __syncthreads();
  for(int o = TPB/2; o > 0; o >>= 1){
    if(threadIdx.x < o) red[threadIdx.x] += red[threadIdx.x+o];
    __syncthreads();
  }
  if(threadIdx.x == 0) atomicAdd(&accum[0], red[0]);
}
__global__ void k_el(const float* __restrict__ sg, float* __restrict__ accum, int n){
  int i = blockIdx.x*blockDim.x + threadIdx.x;
  float s = 0.f;
  if(i < n){
    for(int j = 0; j < 100; ++j){
      float p = sg[(long)i*100 + j];
      s += -p*__logf(p + 1e-15f);
    }
  }
  __shared__ float red[TPB];
  red[threadIdx.x] = s; __syncthreads();
  for(int o = TPB/2; o > 0; o >>= 1){
    if(threadIdx.x < o) red[threadIdx.x] += red[threadIdx.x+o];
    __syncthreads();
  }
  if(threadIdx.x == 0) atomicAdd(&accum[1], red[0]);
}

// ---------------- sparse tmp = A @ sg (never materialize 512MB dense A) ----------------
__global__ void k_tmp_scatter(const int* __restrict__ src, const int* __restrict__ dst,
                              const float* __restrict__ w, const float* __restrict__ sg,
                              float* __restrict__ tmp, long total){
  long t = (long)blockIdx.x*blockDim.x + threadIdx.x;
  if(t >= total) return;
  int e = (int)(t/100), j = (int)(t - (long)e*100);
  atomicAdd(&tmp[(long)src[e]*100 + j], w[e]*sg[(long)dst[e]*100 + j]);
}

// ---------------- xp = sg^T @ x13 per graph (small) ----------------
__global__ void k_xp(const float* __restrict__ sg, const float* __restrict__ x13,
                     float* __restrict__ xp, int B, int N){
  int idx = blockIdx.x*blockDim.x + threadIdx.x;
  if(idx >= B*3000) return;
  int b = idx/3000, r = idx%3000, kc = r/30, d = r%30;
  const float* S = sg  + (long)b*N*100;
  const float* X = x13 + (long)b*N*30;
  float s = 0.f;
  for(int i = 0; i < N; ++i) s += S[(long)i*100 + kc]*X[(long)i*30 + d];
  xp[idx] = s;
}

// ---------------- ap = sg^T @ tmp per graph, WMMA f16->f32 ----------------
// Block = 256 threads (8 waves). Waves 0..6 each own one 16-row strip of the
// 112x112 (padded) output; K staged in LDS in 32-deep chunks as f16.
// LDS layout is [col][k] with row stride 40 halfs (80 B): every per-lane
// fragment slice is 16B-aligned and contiguous -> ds_load_b128 instead of
// 16x ds_load_u16 per fragment.
#define LSTRIDE 40
__global__ void __launch_bounds__(256)
k_ap_wmma(const float* __restrict__ sgp, const float* __restrict__ tmpp,
          float* __restrict__ ap, int N){
  const int K = 100;
  int b = blockIdx.x;
  const float* SG = sgp  + (long)b*N*K;
  const float* TM = tmpp + (long)b*N*K;
  __shared__ __align__(16) _Float16 lsg[112][LSTRIDE];  // lsg[m][k] = sg[kb+k][m]
  __shared__ __align__(16) _Float16 ltm[112][LSTRIDE];  // ltm[n][k] = tmp[kb+k][n]
  int tid  = threadIdx.x;
  int wave = tid >> 5, lane = tid & 31;
  int hl = lane & 15, hi = lane >> 4;

  v8f acc[7];
  v8f zero = {0.f,0.f,0.f,0.f,0.f,0.f,0.f,0.f};
  #pragma unroll
  for(int i = 0; i < 7; ++i) acc[i] = zero;

  int nch = (N + 31)/32;
  for(int ch = 0; ch < nch; ++ch){
    int kb = ch*32;
    __syncthreads();
    for(int idx = tid; idx < 32*112; idx += blockDim.x){
      int kk = idx/112, m = idx - kk*112;      // consecutive tid -> consecutive m: coalesced global
      int gk = kb + kk;
      bool in = (gk < N) && (m < K);
      lsg[m][kk] = (_Float16)(in ? SG[(long)gk*K + m] : 0.f);
      ltm[m][kk] = (_Float16)(in ? TM[(long)gk*K + m] : 0.f);
    }
    __syncthreads();
    if(wave < 7){
      int tm = wave;
      // A fragment (A = sg^T): lanes 0-15 hold K{0..7,16..23}, lanes 16-31 K{8..15,24..31}
      v8h a0 = *(const v8h*)&lsg[tm*16 + hl][hi*8];
      v8h a1 = *(const v8h*)&lsg[tm*16 + hl][16 + hi*8];
      v16h A = __builtin_shufflevector(a0, a1, 0,1,2,3,4,5,6,7,8,9,10,11,12,13,14,15);
      #pragma unroll
      for(int tn = 0; tn < 7; ++tn){
        // B fragment: lanes 0-15 hold K{0..7,16..23}, lanes 16-31 K{8..15,24..31}
        v8h b0 = *(const v8h*)&ltm[tn*16 + hl][hi*8];
        v8h b1 = *(const v8h*)&ltm[tn*16 + hl][16 + hi*8];
        v16h Bf = __builtin_shufflevector(b0, b1, 0,1,2,3,4,5,6,7,8,9,10,11,12,13,14,15);
        acc[tn] = __builtin_amdgcn_wmma_f32_16x16x32_f16(
            false, A, false, Bf, (short)0, acc[tn], false, false);
      }
    }
  }
  if(wave < 7){
    int tm = wave;
    #pragma unroll
    for(int tn = 0; tn < 7; ++tn){
      #pragma unroll
      for(int r = 0; r < 8; ++r){
        int M  = tm*16 + r + hi*8;             // C/D layout: VGPR r, lanes 16-31 = M+8
        int Nc = tn*16 + hl;
        if(M < K && Nc < K) ap[((long)b*K + M)*K + Nc] = acc[tn][r];
      }
    }
  }
}

// ---------------- stage-2 dense GCN on pooled adjacency ----------------
__global__ void k_dih(const float* __restrict__ ap, float* __restrict__ dih, int B){
  int idx = blockIdx.x*blockDim.x + threadIdx.x;
  if(idx >= B*100) return;
  int b = idx/100, i = idx%100;
  float s = 1.0f; // identity on diagonal
  for(int j = 0; j < 100; ++j) s += ap[((long)b*100 + i)*100 + j];
  dih[idx] = rsqrtf(s);
}
__global__ void k_An(const float* __restrict__ ap, const float* __restrict__ dih,
                     float* __restrict__ An, int B){
  long idx = (long)blockIdx.x*blockDim.x + threadIdx.x;
  if(idx >= (long)B*10000) return;
  int b = (int)(idx/10000), r = (int)(idx%10000), i = r/100, j = r%100;
  float a = ap[idx] + ((i == j) ? 1.0f : 0.0f);
  An[idx] = dih[b*100 + i]*a*dih[b*100 + j];
}
__global__ void k_dgcn(const float* __restrict__ An, const float* __restrict__ hw,
                       const float* __restrict__ bias, float* __restrict__ out, int B){
  int idx = blockIdx.x*blockDim.x + threadIdx.x;
  if(idx >= B*3000) return;
  int b = idx/3000, r = idx%3000, kc = r/30, d = r%30;
  float s = bias[d];
  for(int m = 0; m < 100; ++m) s += An[((long)b*100 + kc)*100 + m]*hw[((long)b*100 + m)*30 + d];
  out[idx] = s;
}

// ---------------- final MLP + loss ----------------
__global__ void k_final(const float* __restrict__ x1o, const float* __restrict__ x2o,
                        const float* __restrict__ Wf1, const float* __restrict__ bf1,
                        const float* __restrict__ Wf2, const float* __restrict__ bf2,
                        const float* __restrict__ accum, float* __restrict__ out,
                        int B, int n, int Etot){
  __shared__ float conv[180];
  __shared__ float hid[50];
  int b = blockIdx.x;
  for(int i = threadIdx.x; i < 180; i += blockDim.x)
    conv[i] = (i < 90) ? x1o[b*90 + i] : x2o[b*90 + (i-90)];
  __syncthreads();
  for(int j = threadIdx.x; j < 50; j += blockDim.x){
    float s = bf1[j];
    for(int k = 0; k < 180; ++k) s += conv[k]*Wf1[k*50 + j];
    hid[j] = fmaxf(s, 0.f);
  }
  __syncthreads();
  for(int j = threadIdx.x; j < 6; j += blockDim.x){
    float s = bf2[j];
    for(int k = 0; k < 50; ++k) s += hid[k]*Wf2[k*6 + j];
    out[b*6 + j] = s;
  }
  if(b == 0 && threadIdx.x == 0){
    float ml = -accum[0]/(float)Etot;
    float el =  accum[1]/(float)n;
    out[B*6] = el + ml;
  }
}

// ---------------- host orchestration ----------------
#define LAUNCH(k, total, ...) \
  k<<<dim3((unsigned)(((long)(total) + TPB - 1)/TPB)), dim3(TPB), 0, stream>>>(__VA_ARGS__)

extern "C" void kernel_launch(void* const* d_in, const int* in_sizes, int n_in,
                              void* d_out, int out_size, void* d_ws, size_t ws_size,
                              hipStream_t stream) {
  const float* x    = (const float*)d_in[0];
  const float* ea   = (const float*)d_in[1];
  const float* W11  = (const float*)d_in[2];  const float* b11 = (const float*)d_in[3];
  const float* W12  = (const float*)d_in[4];  const float* b12 = (const float*)d_in[5];
  const float* W13  = (const float*)d_in[6];  const float* b13 = (const float*)d_in[7];
  const float* bn1g = (const float*)d_in[8];  const float* bn1b= (const float*)d_in[9];
  const float* betas= (const float*)d_in[10];
  const float* Wp   = (const float*)d_in[11]; const float* bpb = (const float*)d_in[12];
  const float* W21  = (const float*)d_in[13]; const float* b21 = (const float*)d_in[14];
  const float* W22  = (const float*)d_in[15]; const float* b22 = (const float*)d_in[16];
  const float* W23  = (const float*)d_in[17]; const float* b23 = (const float*)d_in[18];
  const float* bn2g = (const float*)d_in[19]; const float* bn2b= (const float*)d_in[20];
  const float* Wf1  = (const float*)d_in[21]; const float* bf1 = (const float*)d_in[22];
  const float* Wf2  = (const float*)d_in[23]; const float* bf2 = (const float*)d_in[24];
  const int*  eidx  = (const int*)d_in[25];

  const int n    = in_sizes[0]/3;     // 64000
  const int Etot = in_sizes[1];       // 256000 directed edge slots
  const int B    = 32;                // num_graphs (fixed by reference setup)
  const int N    = n/B;               // 2000
  const int* src = eidx;
  const int* dst = eidx + Etot;
  float* out = (float*)d_out;

  // ---- carve workspace ----
  char* base = (char*)d_ws; size_t cur = 0;
  auto alloc = [&](size_t bytes)->float*{
    float* p = (float*)(base + cur);
    cur += (bytes + 255) & ~(size_t)255;
    return p;
  };
  const size_t nf = (size_t)n;
  float* deg   = alloc(nf*4);
  float* dinv  = alloc(nf*4);
  float* enorm = alloc((size_t)Etot*4);
  float* hw    = alloc(nf*30*4);
  float* acc   = alloc(nf*30*4);
  float* x11   = alloc(nf*30*4);
  float* x12   = alloc(nf*30*4);
  float* x13   = alloc(nf*30*4);
  float* stat  = alloc(64*4);
  float* x1o   = alloc((size_t)B*90*4);
  float* bpc   = alloc((size_t)Etot*4);
  float* mu    = alloc((size_t)Etot*8*4);
  float* tbuf  = alloc((size_t)Etot*8*4);
  float* fld   = alloc(nf*8*4);
  float* s1    = alloc(nf*35*4);
  float* sg    = alloc(nf*100*4);
  float* tmp   = alloc(nf*100*4);
  float* xp    = alloc((size_t)B*3000*4);
  float* ap    = alloc((size_t)B*10000*4);
  float* An    = alloc((size_t)B*10000*4);
  float* dih   = alloc((size_t)B*100*4);
  float* hwp   = alloc((size_t)B*3000*4);
  float* x21   = alloc((size_t)B*3000*4);
  float* x22   = alloc((size_t)B*3000*4);
  float* x23   = alloc((size_t)B*3000*4);
  float* x2o   = alloc((size_t)B*90*4);
  float* accum = alloc(2*4);
  (void)n_in; (void)out_size; (void)ws_size;

  // ---- GCN normalization ----
  hipMemsetAsync(deg, 0, nf*4, stream);
  LAUNCH(k_deg_scatter, Etot, dst, ea, deg, Etot);
  LAUNCH(k_dinv, n, deg, dinv, n);
  LAUNCH(k_enorm, Etot, src, dst, ea, dinv, enorm, Etot);

  // ---- stage-1 GCN + BN, 3 layers ----
  auto gcn_layer = [&](const float* hin, int din, const float* W, const float* bias,
                       const float* g, const float* bb, float* xout){
    LAUNCH(k_mm_bias, nf*30, hin, W, (const float*)nullptr, hw, n, din, 30);
    hipMemsetAsync(acc, 0, nf*30*4, stream);
    LAUNCH(k_gcn_scatter, (long)Etot*30, src, dst, enorm, hw, acc, Etot, 30);
    LAUNCH(k_gcn_finish, nf*30, acc, hw, dinv, bias, xout, n, 30);
    k_colstat<<<30, TPB, 0, stream>>>(xout, stat, n, 30);
    LAUNCH(k_bn_apply, nf*30, xout, stat, g, bb, n, 30);
  };
  gcn_layer(x,   3,  W11, b11, bn1g +  0, bn1b +  0, x11);
  gcn_layer(x11, 30, W12, b12, bn1g + 30, bn1b + 30, x12);
  gcn_layer(x12, 30, W13, b13, bn1g + 60, bn1b + 60, x13);
  LAUNCH(k_pool, B*90, x11, x12, x13, x1o, B, N, 30);

  // ---- belief propagation: 7 modules, 10 iters ----
  const int qs[7]   = {2, 3, 4, 5, 6, 7, 8};
  const int offs[7] = {0, 2, 5, 9, 14, 20, 27};
  for(int mi = 0; mi < 7; ++mi){
    int q = qs[mi];
    LAUNCH(k_bp_c, Etot, betas, mi, ea, bpc, Etot);
    LAUNCH(k_bp_init, Etot, mu, Etot, q, mi);
    for(int it = 0; it < 10; ++it){
      hipMemsetAsync(fld, 0, nf*q*4, stream);
      LAUNCH(k_bp_tfld, Etot, dst, bpc, mu, tbuf, fld, Etot, q);
      LAUNCH(k_bp_update, Etot, src, tbuf, fld, mu, Etot, q);
    }
    hipMemsetAsync(fld, 0, nf*q*4, stream);
    LAUNCH(k_bp_tfld, Etot, dst, bpc, mu, tbuf, fld, Etot, q);
    LAUNCH(k_bp_marg, n, fld, s1, n, q, offs[mi]);
  }

  // ---- pool_fc1, softmax (ssm == sg), losses ----
  LAUNCH(k_mm_bias, nf*100, s1, Wp, bpb, sg, n, 35, 100);
  LAUNCH(k_rowsoftmax, n, sg, n, 100);
  hipMemsetAsync(accum, 0, 8, stream);
  LAUNCH(k_ml, Etot, src, dst, sg, accum, Etot);
  LAUNCH(k_el, n, sg, accum, n);

  // ---- sparse tmp = A @ sg (dense A never materialized), xp, ap (WMMA) ----
  hipMemsetAsync(tmp, 0, nf*100*4, stream);
  LAUNCH(k_tmp_scatter, (long)Etot*100, src, dst, ea, sg, tmp, (long)Etot*100);
  LAUNCH(k_xp, B*3000, sg, x13, xp, B, N);
  k_ap_wmma<<<B, 256, 0, stream>>>(sg, tmp, ap, N);

  // ---- stage-2 dense GCN + BN, 3 layers ----
  LAUNCH(k_dih, B*100, ap, dih, B);
  LAUNCH(k_An, (long)B*10000, ap, dih, An, B);
  auto dgcn_layer = [&](const float* hin, const float* W, const float* bias,
                        const float* g, const float* bb, float* xout){
    LAUNCH(k_mm_bias, (long)B*3000, hin, W, (const float*)nullptr, hwp, B*100, 30, 30);
    LAUNCH(k_dgcn, B*3000, An, hwp, bias, xout, B);
    k_colstat<<<30, TPB, 0, stream>>>(xout, stat, B*100, 30);
    LAUNCH(k_bn_apply, (long)B*3000, xout, stat, g, bb, B*100, 30);
  };
  dgcn_layer(xp,  W21, b21, bn2g +  0, bn2b +  0, x21);
  dgcn_layer(x21, W22, b22, bn2g + 30, bn2b + 30, x22);
  dgcn_layer(x22, W23, b23, bn2g + 60, bn2b + 60, x23);
  LAUNCH(k_pool, B*90, x21, x22, x23, x2o, B, 100, 30);

  // ---- final MLP + scalar loss ----
  k_final<<<B, 64, 0, stream>>>(x1o, x2o, Wf1, bf1, Wf2, bf2, accum, out, B, n, Etot);
}
